// JunctionCountsModel_17523466568093
// MI455X (gfx1250) — matrix operand order repaired and checked by reading, not compiled, not run on hardware
//
#include <hip/hip_runtime.h>
#include <math.h>

typedef __attribute__((ext_vector_type(16))) _Float16 v16h;
typedef __attribute__((ext_vector_type(8)))  _Float16 v8h;
typedef __attribute__((ext_vector_type(8)))  float    v8f;

#define L_DIM   32768
#define K_DIM   128
#define N_JUNC  200000
#define H_DIM   768
#define T_DIM   32

#define KT1 9      // 288 / 32 k-tiles for GEMM1
#define NT1 48     // 768 / 16 n-tiles for GEMM1
#define KT2 24     // 768 / 32 k-tiles for GEMM2
#define NT2 2      // 32  / 16 n-tiles for GEMM2

#define AST 296    // LDS A row stride in halves (288 + 8 pad, conflict-free)
#define YST 776    // LDS Y row stride in halves (768 + 8 pad, conflict-free)
#define MB  32     // junctions per block (2 WMMA M-tiles)

// ---------------------------------------------------------------------------
// One-time weight repack: f32 weights -> f16 WMMA B-fragments in workspace.
// B fragment layout (32x16 f16, wave32): lane c<16 holds column c, K rows
// kt*32+0..15; lane c>=16 holds column c-16, K rows kt*32+16..31. Each
// lane's 16 halves stored contiguously -> main kernel does 32B loads.
// ---------------------------------------------------------------------------
__global__ void pack_weights(const float* __restrict__ Wd,
                             const float* __restrict__ Wa,
                             const float* __restrict__ wdist,
                             const float* __restrict__ Wde,
                             const float* __restrict__ Wout,
                             _Float16* __restrict__ Wp1,
                             _Float16* __restrict__ Wp2) {
  int gid = blockIdx.x * blockDim.x + threadIdx.x;
  if (gid < NT1 * KT1 * 32) {
    int lane = gid & 31;
    int fi   = gid >> 5;
    int nt   = fi / KT1, kt = fi % KT1;
    int col   = nt * 16 + (lane & 15);
    int kbase = kt * 32 + ((lane < 16) ? 0 : 16);
    _Float16* dst = Wp1 + (size_t)gid * 16;
#pragma unroll
    for (int j = 0; j < 16; ++j) {
      int k = kbase + j;
      float v;
      if (k < 128)       v = Wd[(size_t)k * H_DIM + col];
      else if (k < 256)  v = Wa[(size_t)(k - 128) * H_DIM + col];
      else if (k < 271)  v = Wde[(size_t)(k - 256) * H_DIM + col];
      else if (k == 271) v = wdist[col];
      else               v = 0.0f;
      dst[j] = (_Float16)v;
    }
  } else {
    int gi = gid - NT1 * KT1 * 32;
    if (gi >= NT2 * KT2 * 32) return;
    int lane = gi & 31;
    int fi   = gi >> 5;
    int nt   = fi / KT2, kt = fi % KT2;
    int col   = nt * 16 + (lane & 15);
    int kbase = kt * 32 + ((lane < 16) ? 0 : 16);
    _Float16* dst = Wp2 + (size_t)gi * 16;
#pragma unroll
    for (int j = 0; j < 16; ++j)
      dst[j] = (_Float16)Wout[(size_t)(kbase + j) * T_DIM + col];
  }
}

// ---------------------------------------------------------------------------
// Branch-free transcendentals on the CDNA5 TRANS32 pipe (co-issues with WMMA)
// ---------------------------------------------------------------------------
#define LOG2E_F 1.4426950408889634f
#define LN2_F   0.6931471805599453f

__device__ __forceinline__ float fast_tanh(float z) {
#if __has_builtin(__builtin_amdgcn_tanhf)
  return __builtin_amdgcn_tanhf(z);              // v_tanh_f32
#else
  // tanh(z) = 1 - 2/(exp(2z)+1), via v_exp_f32 (base 2) + v_rcp_f32
  float e = __builtin_amdgcn_exp2f(z * (2.0f * LOG2E_F));
  return 1.0f - 2.0f * __builtin_amdgcn_rcpf(e + 1.0f);
#endif
}

__device__ __forceinline__ float gelu_f(float x) {
  const float c = 0.7978845608028654f; // sqrt(2/pi)
  float t = fast_tanh(c * (x + 0.044715f * x * x * x));
  return 0.5f * x * (1.0f + t);
}

__device__ __forceinline__ float softplus_f(float x) {
  // log1p(exp(x)) = max(x,0) + ln2 * log2(1 + exp2(-|x|*log2e)); branch-free
  float e = __builtin_amdgcn_exp2f(-fabsf(x) * LOG2E_F);   // v_exp_f32
  float l = __builtin_amdgcn_logf(1.0f + e);               // v_log_f32 (log2)
  return fmaxf(x, 0.0f) + LN2_F * l;
}

// A-fragment (16x32 f16): lane l holds row l%16; lanes 0-15 take K = 0..15,
// lanes 16-31 take K = 16..31 (two ds_load_b128 per frag).
__device__ __forceinline__ v16h load_a_frag(const _Float16* row, int kt, int lane) {
  int kb = (lane < 16) ? 0 : 16;
  v8h lo = *(const v8h*)(row + kt * 32 + kb);
  v8h hi = *(const v8h*)(row + kt * 32 + kb + 8);
  return __builtin_shufflevector(lo, hi, 0,1,2,3,4,5,6,7,8,9,10,11,12,13,14,15);
}

__global__ void __launch_bounds__(128)
junction_kernel(const float* __restrict__ x,
                const int* __restrict__ xxj,
                const float* __restrict__ b_in,
                const float* __restrict__ b_out,
                const _Float16* __restrict__ Wp1,
                const _Float16* __restrict__ Wp2,
                float* __restrict__ out) {
  extern __shared__ _Float16 smem[];
  _Float16* As = smem;             // [MB][AST] gathered A-tile
  _Float16* Ys = smem + MB * AST;  // [MB][YST] post-GELU hidden tile

  const int tid  = threadIdx.x;
  const int lane = tid & 31;
  const int wave = tid >> 5;
  const size_t jblock = (size_t)blockIdx.x * MB;

  // ---- Stage 0: gather x rows, convert f32->f16, build dist features ----
  {
    int j = tid >> 2;   // junction within block: 0..31
    int t = tid & 3;    // quarter of the K=128 row
    const int* idx = xxj + (jblock + j) * 3;
    int bi = idx[0], di = idx[1], ai = idx[2];
    const float4* xd = (const float4*)(x + ((size_t)bi * L_DIM + (size_t)di) * K_DIM);
    const float4* xa = (const float4*)(x + ((size_t)bi * L_DIM + (size_t)ai) * K_DIM);
    _Float16* arow = As + j * AST;
#pragma unroll
    for (int q = 0; q < 4; ++q) {
      float4 f0 = xd[t * 8 + 2 * q], f1 = xd[t * 8 + 2 * q + 1];
      v8h h = { (_Float16)f0.x, (_Float16)f0.y, (_Float16)f0.z, (_Float16)f0.w,
                (_Float16)f1.x, (_Float16)f1.y, (_Float16)f1.z, (_Float16)f1.w };
      *(v8h*)(arow + t * 32 + q * 8) = h;
    }
#pragma unroll
    for (int q = 0; q < 4; ++q) {
      float4 f0 = xa[t * 8 + 2 * q], f1 = xa[t * 8 + 2 * q + 1];
      v8h h = { (_Float16)f0.x, (_Float16)f0.y, (_Float16)f0.z, (_Float16)f0.w,
                (_Float16)f1.x, (_Float16)f1.y, (_Float16)f1.z, (_Float16)f1.w };
      *(v8h*)(arow + 128 + t * 32 + q * 8) = h;
    }
    if (t == 0) {
      int dist = ai - di; if (dist < 0) dist = -dist;
#pragma unroll
      for (int d = 0; d < 15; ++d)
        arow[256 + d] = (_Float16)((dist >= (1 << d)) ? 1.0f : 0.0f);
      arow[271] = (_Float16)((float)dist * (1.0f / 32768.0f)); // dist / L
#pragma unroll
      for (int d = 272; d < 288; ++d)
        arow[d] = (_Float16)0.0f;                               // K padding
    }
  }
  __syncthreads();

  const int colInTile = lane & 15;
  const int rbase     = (lane >= 16) ? 8 : 0;
  const _Float16* arow0 = As + colInTile * AST;         // M-tile 0 frag row
  const _Float16* arow1 = As + (16 + colInTile) * AST;  // M-tile 1 frag row

  // ---- Stage 1: y = A(32x288) @ W''(288x768), +b_in, GELU -> Ys (f16) ----
  for (int i = 0; i < NT1 / 4; ++i) {
    int nt = wave * (NT1 / 4) + i;
    v8f acc0 = {}; v8f acc1 = {};
    const v16h* bbase = (const v16h*)Wp1 + (size_t)nt * KT1 * 32 + lane;
    for (int kt = 0; kt < KT1; ++kt) {
      v16h b  = bbase[kt * 32];
      v16h a0 = load_a_frag(arow0, kt, lane);
      v16h a1 = load_a_frag(arow1, kt, lane);
      acc0 = __builtin_amdgcn_wmma_f32_16x16x32_f16(false, a0, false, b,
                                                    (short)0, acc0, false, false);
      acc1 = __builtin_amdgcn_wmma_f32_16x16x32_f16(false, a1, false, b,
                                                    (short)0, acc1, false, false);
    }
    int h = nt * 16 + colInTile;
    float bin = b_in[h];
#pragma unroll
    for (int r = 0; r < 8; ++r) {
      Ys[(rbase + r) * YST + h]      = (_Float16)gelu_f(acc0[r] + bin);
      Ys[(16 + rbase + r) * YST + h] = (_Float16)gelu_f(acc1[r] + bin);
    }
  }
  __syncthreads();

  // ---- Stage 2: out = softplus(Ys(32x768) @ W_out(768x32) + b_out) ----
  {
    int mtile = wave >> 1;
    int nt    = wave & 1;
    const _Float16* yrow = Ys + (mtile * 16 + colInTile) * YST;
    v8f acc = {};
    const v16h* bbase = (const v16h*)Wp2 + (size_t)nt * KT2 * 32 + lane;
    for (int kt = 0; kt < KT2; ++kt) {
      v16h b = bbase[kt * 32];
      v16h a = load_a_frag(yrow, kt, lane);
      acc = __builtin_amdgcn_wmma_f32_16x16x32_f16(false, a, false, b,
                                                   (short)0, acc, false, false);
    }
    int t = nt * 16 + colInTile;
    float bo = b_out[t];
#pragma unroll
    for (int r = 0; r < 8; ++r) {
      size_t n = jblock + (size_t)(mtile * 16 + rbase + r);
      out[n * T_DIM + t] = softplus_f(acc[r] + bo);
    }
  }
}

// ---------------------------------------------------------------------------
extern "C" void kernel_launch(void* const* d_in, const int* in_sizes, int n_in,
                              void* d_out, int out_size, void* d_ws, size_t ws_size,
                              hipStream_t stream) {
  (void)in_sizes; (void)n_in; (void)out_size; (void)ws_size;
  const float* x     = (const float*)d_in[0];
  const int*   xxj   = (const int*)  d_in[1];
  const float* Wd    = (const float*)d_in[2];
  const float* Wa    = (const float*)d_in[3];
  const float* wdist = (const float*)d_in[4];
  const float* Wde   = (const float*)d_in[5];
  const float* b_in  = (const float*)d_in[6];
  const float* Wout  = (const float*)d_in[7];
  const float* b_out = (const float*)d_in[8];
  float* out = (float*)d_out;

  _Float16* Wp1 = (_Float16*)d_ws;
  _Float16* Wp2 = (_Float16*)((char*)d_ws + (size_t)NT1 * KT1 * 32 * 16 * 2);

  int packThreads = (NT1 * KT1 + NT2 * KT2) * 32; // 15360
  pack_weights<<<(packThreads + 255) / 256, 256, 0, stream>>>(
      Wd, Wa, wdist, Wde, Wout, Wp1, Wp2);

  size_t lds = (size_t)(MB * AST + MB * YST) * sizeof(_Float16); // ~67 KB
  junction_kernel<<<N_JUNC / MB, 128, lds, stream>>>(
      x, xxj, b_in, b_out, Wp1, Wp2, out);
}
